// ZoomFormer_51659866636377
// MI455X (gfx1250) — matrix operand order repaired and checked by reading, not compiled, not run on hardware
//
#include <hip/hip_runtime.h>

#define DIM   512
#define HEADS 8
#define HD    64
#define NKV   20
#define BTOK  3136   // 64*7*7
#define BATCH 64

typedef unsigned short ushort_t;
typedef __attribute__((ext_vector_type(16))) __bf16 v16bf;
typedef __attribute__((ext_vector_type(8)))  float  v8f;

static_assert(sizeof(v16bf) == 32, "v16bf must be 32 bytes");

__device__ __forceinline__ float bf2f(ushort_t h) {
  return __uint_as_float(((unsigned int)h) << 16);
}
__device__ __forceinline__ ushort_t f2bf(float f) {
  unsigned int u = __float_as_uint(f);
  u += 0x7FFFu + ((u >> 16) & 1u);   // round-to-nearest-even
  return (ushort_t)(u >> 16);
}

union FragU { uint4 q[2]; v16bf v; };

struct ConvRow { int b, y, x; };

// Fragment set for one 32-K step of a 32M x 64N wave tile
struct Frags {
  FragU a[2];   // rows [m0, m0+16) and [m0+16, m0+32)
  FragU b[4];   // cols n0 + t*16
};

// ---------------------------------------------------------------------------
// Fragment loaders. A/B both K-contiguous bf16.
// A layout per ISA (16-bit A 16x32): lane<16 holds K [0..7]+[16..23] of row
// (lane&15); lane>=16 holds K [8..15]+[24..31] of the same row.
// B layout: lane<16 holds K [0..7]+[16..23] of col (lane&15); lane>=16 holds K [8..15]+[24..31].
// ---------------------------------------------------------------------------
template<bool ACONV>
__device__ __forceinline__ void load_frags(Frags& f,
    const ushort_t* __restrict__ A, const ushort_t* __restrict__ Bw,
    int m0, int n0, int K, int logC, int k0,
    int r, int hs, int kb, const ConvRow cr[2])
{
  if (!ACONV) {
    #pragma unroll
    for (int u = 0; u < 2; ++u) {
      const ushort_t* ap = A + (size_t)(m0 + u * 16 + r) * K + k0;
      f.a[u].q[0] = *(const uint4*)(ap + hs);
      f.a[u].q[1] = *(const uint4*)(ap + 16 + hs);
    }
  } else {
    const int Cm1 = (1 << logC) - 1;
    int base0 = k0 + hs;
    int s0 = base0 >> logC, i0 = base0 & Cm1;
    int base1 = k0 + 16 + hs;
    int s1 = base1 >> logC, i1 = base1 & Cm1;
    int dy0 = s0 / 3, dx0 = s0 % 3;
    int dy1 = s1 / 3, dx1 = s1 % 3;
    #pragma unroll
    for (int u = 0; u < 2; ++u) {
      f.a[u].q[0] = *(const uint4*)(A +
          (((size_t)(cr[u].b * 81 + (cr[u].y + dy0) * 9 + cr[u].x + dx0)) << logC) + i0);
      f.a[u].q[1] = *(const uint4*)(A +
          (((size_t)(cr[u].b * 81 + (cr[u].y + dy1) * 9 + cr[u].x + dx1)) << logC) + i1);
    }
  }
  #pragma unroll
  for (int t = 0; t < 4; ++t) {
    const ushort_t* bp = Bw + (size_t)(n0 + t * 16 + r) * K + k0 + hs;
    f.b[t].q[0] = *(const uint4*)(bp);
    f.b[t].q[1] = *(const uint4*)(bp + 16);
  }
}

__device__ __forceinline__ void do_mma(v8f acc[2][4], const Frags& f)
{
  #pragma unroll
  for (int u = 0; u < 2; ++u)
    #pragma unroll
    for (int t = 0; t < 4; ++t)
      acc[u][t] = __builtin_amdgcn_wmma_f32_16x16x32_bf16(
          false, f.a[u].v, false, f.b[t].v, (short)0, acc[u][t], false, false);
}

// ---------------------------------------------------------------------------
// Generic bf16 WMMA GEMM:  C[M,N] = A[M,K] * Bw[N,K]^T   (both K-contiguous)
// ACONV=true: A is an implicit-GEMM patch read from a padded activation
//             tensor laid out [b][9][9][C] with K ordered as (s=ky*3+kx, i).
// EPI: 0=f32 store, 1=bf16 store, 2=bias+LeakyReLU -> padded bf16 store,
//      3=bias + residual add into `resid` (in place).
// Wave tile 32Mx64N, software-pipelined (ping-pong register double buffer).
// Block = 8 waves as 2(M) x 4(N) -> 64M x 256N. Requires K % 64 == 0.
// ---------------------------------------------------------------------------
template<bool ACONV, int EPI>
__global__ __launch_bounds__(256)
void gemm_bf16_wmma(const ushort_t* __restrict__ A,
                    const ushort_t* __restrict__ Bw,
                    int M, int N, int K, int logC,
                    float* __restrict__ outF,
                    ushort_t* __restrict__ outB,
                    const float* __restrict__ bias,
                    float* __restrict__ resid)
{
  const int lane = threadIdx.x & 31;
  const int wid  = threadIdx.x >> 5;
  const int wm   = wid & 1;
  const int wn   = wid >> 1;
  const int m0   = blockIdx.x * 64 + wm * 32;
  const int n0   = blockIdx.y * 256 + wn * 64;

  const int r  = lane & 15;
  const int hs = (lane >> 4) << 3;   // A: 0 or 8
  const int kb = (lane >> 4) << 4;   // unused; B also uses hs (0 or 8)

  v8f acc[2][4] = {};

  ConvRow cr[2] = {};
  if (ACONV) {
    #pragma unroll
    for (int u = 0; u < 2; ++u) {
      int m = m0 + u * 16 + r;
      cr[u].b = m / 49;
      int rr = m - cr[u].b * 49;
      cr[u].y = rr / 7;
      cr[u].x = rr - cr[u].y * 7;
    }
  }

  Frags cur, nxt;
  load_frags<ACONV>(cur, A, Bw, m0, n0, K, logC, 0, r, hs, kb, cr);

  for (int k0 = 0; k0 < K; k0 += 64) {
    // prefetch second half-step while first computes
    load_frags<ACONV>(nxt, A, Bw, m0, n0, K, logC, k0 + 32, r, hs, kb, cr);
    do_mma(acc, cur);
    if (k0 + 64 < K)
      load_frags<ACONV>(cur, A, Bw, m0, n0, K, logC, k0 + 64, r, hs, kb, cr);
    do_mma(acc, nxt);
  }

  const int mhalf = (lane >> 4) << 3;
  #pragma unroll
  for (int u = 0; u < 2; ++u) {
    const int mbase = m0 + u * 16 + mhalf;
    #pragma unroll
    for (int t = 0; t < 4; ++t) {
      const int n = n0 + t * 16 + r;
      #pragma unroll
      for (int j = 0; j < 8; ++j) {
        const int m = mbase + j;
        float v = acc[u][t][j];
        if constexpr (EPI == 0) {
          outF[(size_t)m * N + n] = v;
        } else if constexpr (EPI == 1) {
          outB[(size_t)m * N + n] = f2bf(v);
        } else if constexpr (EPI == 2) {
          v += bias[n];
          v = v > 0.f ? v : 0.01f * v;
          int b = m / 49; int rr = m - b * 49; int y = rr / 7; int x = rr - y * 7;
          outB[(size_t)(b * 81 + (y + 1) * 9 + (x + 1)) * N + n] = f2bf(v);
        } else {
          size_t idx = (size_t)m * N + n;
          resid[idx] = resid[idx] + v + bias[n];
        }
      }
    }
  }
}

// ---------------------------------------------------------------------------
// Attention: one wave per (token, head). 20 context tokens, head dim 64.
// ---------------------------------------------------------------------------
__global__ __launch_bounds__(256)
void attn_kernel(const float* __restrict__ qbuf, const ushort_t* __restrict__ kv,
                 ushort_t* __restrict__ obuf)
{
  const int m    = blockIdx.x;
  const int h    = threadIdx.x >> 5;
  const int lane = threadIdx.x & 31;
  const int d0   = lane * 2;

  const float* qp = qbuf + (size_t)m * DIM + h * HD;
  const float q0 = qp[d0], q1 = qp[d0 + 1];
  const ushort_t* kvb = kv + (size_t)m * NKV * (2 * DIM);
  const float scale = 0.04419417382415922f;   // 512^-0.5

  float s[NKV];
  #pragma unroll
  for (int j = 0; j < NKV; ++j) {
    const ushort_t* kp = kvb + (size_t)j * (2 * DIM) + h * HD;
    float d = q0 * bf2f(kp[d0]) + q1 * bf2f(kp[d0 + 1]);
    #pragma unroll
    for (int o = 16; o; o >>= 1) d += __shfl_xor(d, o, 32);
    s[j] = d * scale;
  }
  float mx = s[0];
  #pragma unroll
  for (int j = 1; j < NKV; ++j) mx = fmaxf(mx, s[j]);
  float sum = 0.f;
  #pragma unroll
  for (int j = 0; j < NKV; ++j) { s[j] = __expf(s[j] - mx); sum += s[j]; }
  const float inv = 1.f / sum;

  float o0 = 0.f, o1 = 0.f;
  #pragma unroll
  for (int j = 0; j < NKV; ++j) {
    const ushort_t* vp = kvb + (size_t)j * (2 * DIM) + DIM + h * HD;
    const float a = s[j] * inv;
    o0 += a * bf2f(vp[d0]);
    o1 += a * bf2f(vp[d0 + 1]);
  }
  obuf[(size_t)m * DIM + h * HD + d0]     = f2bf(o0);
  obuf[(size_t)m * DIM + h * HD + d0 + 1] = f2bf(o1);
}

__device__ __forceinline__ float blockSum(float v, float* sm) {
  #pragma unroll
  for (int o = 16; o; o >>= 1) v += __shfl_xor(v, o, 32);
  __syncthreads();
  if ((threadIdx.x & 31) == 0) sm[threadIdx.x >> 5] = v;
  __syncthreads();
  float r = 0.f;
  #pragma unroll
  for (int i = 0; i < 8; ++i) r += sm[i];
  return r;
}

// xq += LayerNorm(oproj + out_b) * g + b   (per-token over 512 channels)
__global__ __launch_bounds__(256)
void ln_residual_kernel(const float* __restrict__ oproj, const float* __restrict__ ob,
                        const float* __restrict__ g, const float* __restrict__ be,
                        float* __restrict__ xq)
{
  __shared__ float sm[8];
  const int m = blockIdx.x, t = threadIdx.x;
  float v0 = oproj[(size_t)m * DIM + t]       + ob[t];
  float v1 = oproj[(size_t)m * DIM + t + 256] + ob[t + 256];
  float s  = blockSum(v0 + v1, sm);
  float sq = blockSum(v0 * v0 + v1 * v1, sm);
  float mean = s * (1.f / DIM);
  float var  = sq * (1.f / DIM) - mean * mean;
  float rstd = rsqrtf(var + 1e-5f);
  xq[(size_t)m * DIM + t]       += (v0 - mean) * rstd * g[t]       + be[t];
  xq[(size_t)m * DIM + t + 256] += (v1 - mean) * rstd * g[t + 256] + be[t + 256];
}

// Channel norm (note: EPS added to sqrt) -> padded bf16 activation [b][9][9][512]
__global__ __launch_bounds__(256)
void cnorm_kernel(const float* __restrict__ xq, const float* __restrict__ cg,
                  const float* __restrict__ cb, ushort_t* __restrict__ npad)
{
  __shared__ float sm[8];
  const int m = blockIdx.x, t = threadIdx.x;
  float v0 = xq[(size_t)m * DIM + t];
  float v1 = xq[(size_t)m * DIM + t + 256];
  float s  = blockSum(v0 + v1, sm);
  float sq = blockSum(v0 * v0 + v1 * v1, sm);
  float mean = s * (1.f / DIM);
  float var  = fmaxf(sq * (1.f / DIM) - mean * mean, 0.f);
  float rden = 1.f / (sqrtf(var) + 1e-5f);
  int b = m / 49, rr = m - b * 49, y = rr / 7, x = rr - y * 7;
  size_t base = ((size_t)(b * 81 + (y + 1) * 9 + (x + 1))) << 9;
  npad[base + t]       = f2bf((v0 - mean) * rden * cg[t]       + cb[t]);
  npad[base + t + 256] = f2bf((v1 - mean) * rden * cg[t + 256] + cb[t + 256]);
}

// Pack context tokens (t2 then t1) into bf16 [B*20][512]
__global__ void pack_tkv_kernel(const float* __restrict__ x1, const float* __restrict__ x2,
                                ushort_t* __restrict__ tkv, int total)
{
  for (int idx = blockIdx.x * blockDim.x + threadIdx.x; idx < total;
       idx += gridDim.x * blockDim.x) {
    int ch = idx & (DIM - 1);
    int mp = idx >> 9;
    int j  = mp % NKV;
    int tb = mp / NKV;
    int b = tb / 49, rr = tb - b * 49, oh = rr / 7, ow = rr - oh * 7;
    float v;
    if (j < 4) {
      int kh = j >> 1, kw = j & 1;
      v = x2[(((size_t)b * DIM + ch) * 14 + (oh * 2 + kh)) * 14 + (ow * 2 + kw)];
    } else {
      int jj = j - 4; int kh = jj >> 2, kw = jj & 3;
      v = x1[(((size_t)b * DIM + ch) * 28 + (oh * 4 + kh)) * 28 + (ow * 4 + kw)];
    }
    tkv[idx] = f2bf(v);
  }
}

__global__ void init_xq_kernel(const float* __restrict__ x3, float* __restrict__ xq, int total)
{
  for (int idx = blockIdx.x * blockDim.x + threadIdx.x; idx < total;
       idx += gridDim.x * blockDim.x) {
    int ch = idx & (DIM - 1);
    int tb = idx >> 9;
    int b = tb / 49, rr = tb - b * 49, oh = rr / 7, ow = rr - oh * 7;
    xq[idx] = x3[(((size_t)b * DIM + ch) * 7 + oh) * 7 + ow];
  }
}

__global__ void final_out_kernel(const float* __restrict__ xq, float* __restrict__ out, int total)
{
  for (int idx = blockIdx.x * blockDim.x + threadIdx.x; idx < total;
       idx += gridDim.x * blockDim.x) {
    int x = idx % 7; int t1 = idx / 7; int y = t1 % 7; int t2 = t1 / 7;
    int ch = t2 & (DIM - 1); int b = t2 >> 9;
    out[idx] = xq[(((size_t)(b * 49 + y * 7 + x)) << 9) + ch];
  }
}

__global__ void cvt_bf16_kernel(const float* __restrict__ src, ushort_t* __restrict__ dst, int n)
{
  for (int i = blockIdx.x * blockDim.x + threadIdx.x; i < n; i += gridDim.x * blockDim.x)
    dst[i] = f2bf(src[i]);
}

// conv weight (O,I,3,3) -> bf16 [O][s*I + i], s = ky*3+kx
__global__ void repack_conv_kernel(const float* __restrict__ w, ushort_t* __restrict__ dst,
                                   int I, int total)
{
  for (int idx = blockIdx.x * blockDim.x + threadIdx.x; idx < total;
       idx += gridDim.x * blockDim.x) {
    int s = idx % 9; int t = idx / 9; int i = t % I; int o = t / I;
    dst[(size_t)o * I * 9 + s * I + i] = f2bf(w[idx]);
  }
}

__global__ void zero_u16_kernel(ushort_t* __restrict__ p, int n)
{
  for (int i = blockIdx.x * blockDim.x + threadIdx.x; i < n; i += gridDim.x * blockDim.x)
    p[i] = 0;
}

static inline int gblk(int n) { return (n + 255) / 256; }

extern "C" void kernel_launch(void* const* d_in, const int* in_sizes, int n_in,
                              void* d_out, int out_size, void* d_ws, size_t ws_size,
                              hipStream_t stream)
{
  (void)in_sizes; (void)n_in; (void)out_size; (void)ws_size;
  const float* x1    = (const float*)d_in[0];
  const float* x2    = (const float*)d_in[1];
  const float* x3    = (const float*)d_in[2];
  const float* q_w   = (const float*)d_in[3];
  const float* kv_w  = (const float*)d_in[4];
  const float* out_w = (const float*)d_in[5];
  const float* out_b = (const float*)d_in[6];
  const float* ln_g  = (const float*)d_in[7];
  const float* ln_b  = (const float*)d_in[8];
  const float* cln_g = (const float*)d_in[9];
  const float* cln_b = (const float*)d_in[10];
  const float* c1w   = (const float*)d_in[11];
  const float* c1b   = (const float*)d_in[12];
  const float* c2w   = (const float*)d_in[13];
  const float* c2b   = (const float*)d_in[14];

  uintptr_t base = (uintptr_t)d_ws;
  auto alloc = [&](size_t bytes) -> void* {
    uintptr_t p = base;
    base += (bytes + 255) & ~(size_t)255;
    return (void*)p;
  };

  const int MKV = BTOK * NKV;                 // 62720
  ushort_t* Tkv    = (ushort_t*)alloc((size_t)MKV * DIM * 2);
  ushort_t* kvbuf  = (ushort_t*)alloc((size_t)MKV * 2 * DIM * 2);
  float*    xq     = (float*)   alloc((size_t)BTOK * DIM * 4);
  ushort_t* xq_bf  = (ushort_t*)alloc((size_t)BTOK * DIM * 2);
  float*    qbuf   = (float*)   alloc((size_t)BTOK * DIM * 4);
  ushort_t* obuf   = (ushort_t*)alloc((size_t)BTOK * DIM * 2);
  float*    oproj  = (float*)   alloc((size_t)BTOK * DIM * 4);
  ushort_t* qw_bf  = (ushort_t*)alloc((size_t)DIM * DIM * 2);
  ushort_t* kvw_bf = (ushort_t*)alloc((size_t)2 * DIM * DIM * 2);
  ushort_t* ow_bf  = (ushort_t*)alloc((size_t)DIM * DIM * 2);
  ushort_t* w1p    = (ushort_t*)alloc((size_t)1024 * 4608 * 2);
  ushort_t* w2p    = (ushort_t*)alloc((size_t)512 * 9216 * 2);
  ushort_t* npad   = (ushort_t*)alloc((size_t)BATCH * 81 * 512 * 2);
  ushort_t* fpad   = (ushort_t*)alloc((size_t)BATCH * 81 * 1024 * 2);

  // One-time prep: zero padded activations (borders stay zero), pack tokens, init xq
  {
    int nz1 = BATCH * 81 * 512, nz2 = BATCH * 81 * 1024;
    zero_u16_kernel<<<gblk(nz1), 256, 0, stream>>>(npad, nz1);
    zero_u16_kernel<<<gblk(nz2), 256, 0, stream>>>(fpad, nz2);
    int npk = MKV * DIM;
    pack_tkv_kernel<<<gblk(npk), 256, 0, stream>>>(x1, x2, Tkv, npk);
    int nxq = BTOK * DIM;
    init_xq_kernel<<<gblk(nxq), 256, 0, stream>>>(x3, xq, nxq);
  }

  for (int i = 0; i < 2; ++i) {
    const float* qw_i  = q_w   + (size_t)i * DIM * DIM;
    const float* kvw_i = kv_w  + (size_t)i * 2 * DIM * DIM;
    const float* ow_i  = out_w + (size_t)i * DIM * DIM;
    const float* ob_i  = out_b + (size_t)i * DIM;
    const float* lg_i  = ln_g  + (size_t)i * DIM;
    const float* lb_i  = ln_b  + (size_t)i * DIM;
    const float* cg_i  = cln_g + (size_t)i * DIM;
    const float* cb_i  = cln_b + (size_t)i * DIM;
    const float* w1_i  = c1w + (size_t)i * 1024 * 512 * 9;
    const float* b1_i  = c1b + (size_t)i * 1024;
    const float* w2_i  = c2w + (size_t)i * 512 * 1024 * 9;
    const float* b2_i  = c2b + (size_t)i * 512;

    // Weight conversion / repack for this depth
    cvt_bf16_kernel<<<gblk(DIM * DIM), 256, 0, stream>>>(qw_i, qw_bf, DIM * DIM);
    cvt_bf16_kernel<<<gblk(2 * DIM * DIM), 256, 0, stream>>>(kvw_i, kvw_bf, 2 * DIM * DIM);
    cvt_bf16_kernel<<<gblk(DIM * DIM), 256, 0, stream>>>(ow_i, ow_bf, DIM * DIM);
    repack_conv_kernel<<<gblk(1024 * 512 * 9), 256, 0, stream>>>(w1_i, w1p, 512, 1024 * 512 * 9);
    repack_conv_kernel<<<gblk(512 * 1024 * 9), 256, 0, stream>>>(w2_i, w2p, 1024, 512 * 1024 * 9);

    // Q projection: qbuf(f32) = xq(bf16) * q_w^T
    cvt_bf16_kernel<<<gblk(BTOK * DIM), 256, 0, stream>>>(xq, xq_bf, BTOK * DIM);
    gemm_bf16_wmma<false, 0><<<dim3(BTOK / 64, DIM / 256), 256, 0, stream>>>(
        xq_bf, qw_bf, BTOK, DIM, DIM, 0, qbuf, nullptr, nullptr, nullptr);

    // KV projection: kvbuf(bf16) = Tkv * kv_w^T   (the big GEMM)
    gemm_bf16_wmma<false, 1><<<dim3(MKV / 64, (2 * DIM) / 256), 256, 0, stream>>>(
        Tkv, kvw_bf, MKV, 2 * DIM, DIM, 0, nullptr, kvbuf, nullptr, nullptr);

    // Attention -> obuf(bf16)
    attn_kernel<<<BTOK, 256, 0, stream>>>(qbuf, kvbuf, obuf);

    // Out projection: oproj(f32) = obuf * out_w^T
    gemm_bf16_wmma<false, 0><<<dim3(BTOK / 64, DIM / 256), 256, 0, stream>>>(
        obuf, ow_bf, BTOK, DIM, DIM, 0, oproj, nullptr, nullptr, nullptr);

    // LayerNorm(+out_b) + residual into xq
    ln_residual_kernel<<<BTOK, 256, 0, stream>>>(oproj, ob_i, lg_i, lb_i, xq);

    // Channel norm -> padded bf16 activations
    cnorm_kernel<<<BTOK, 256, 0, stream>>>(xq, cg_i, cb_i, npad);

    // Conv1 (implicit GEMM, K = 512*9): fpad = LeakyReLU(conv3x3(npad) + b1)
    gemm_bf16_wmma<true, 2><<<dim3(BTOK / 64, 1024 / 256), 256, 0, stream>>>(
        npad, w1p, BTOK, 1024, 512 * 9, 9, nullptr, fpad, b1_i, nullptr);

    // Conv2 (implicit GEMM, K = 1024*9): xq += conv3x3(fpad) + b2
    gemm_bf16_wmma<true, 3><<<dim3(BTOK / 64, DIM / 256), 256, 0, stream>>>(
        fpad, w2p, BTOK, DIM, 1024 * 9, 10, nullptr, nullptr, b2_i, xq);
  }

  // Emit NCHW fp32 output
  final_out_kernel<<<gblk(BATCH * DIM * 49), 256, 0, stream>>>(xq, (float*)d_out, BATCH * DIM * 49);
}
